// Attention_13503377178798
// MI455X (gfx1250) — matrix operand order repaired and checked by reading, not compiled, not run on hardware
//
#include <hip/hip_runtime.h>
#include <hip/hip_fp16.h>

typedef __attribute__((ext_vector_type(16))) _Float16 v16h;
typedef __attribute__((ext_vector_type(8)))  _Float16 v8h;
typedef __attribute__((ext_vector_type(8)))  float    v8f;

#define S_LEN    2048
#define D_MODEL  4096
#define NH       32
#define NKVH     8
#define HD       128
#define KV_DIM   (NKVH * HD)   // 1024
#define RANK     16
#define LORA_SCALE 2.0f        // 32/16
#define INV_SQRT_HD 0.08838834764831845f

__device__ __constant__ float NF4LUT[16] = {
  -1.0f, -0.6961928009986877f, -0.5250730514526367f, -0.39491748809814453f,
  -0.28444138169288635f, -0.18477343022823334f, -0.09105003625154495f, 0.0f,
  0.07958029955625534f, 0.16093020141124725f, 0.24611230194568634f,
  0.33791524171829224f, 0.44070982933044434f, 0.5626170039176941f,
  0.7229568362236023f, 1.0f };

__device__ __forceinline__ v16h cat16(v8h a, v8h b) {
  return __builtin_shufflevector(a, b, 0,1,2,3,4,5,6,7,8,9,10,11,12,13,14,15);
}
__device__ __forceinline__ v8f wmma_f16(v16h a, v16h b, v8f c) {
  return __builtin_amdgcn_wmma_f32_16x16x32_f16(false, a, false, b, (short)0, c, false, false);
}
__device__ __forceinline__ float rowmax16(float v) {
  v = fmaxf(v, __shfl_xor(v, 1, 32));
  v = fmaxf(v, __shfl_xor(v, 2, 32));
  v = fmaxf(v, __shfl_xor(v, 4, 32));
  v = fmaxf(v, __shfl_xor(v, 8, 32));
  return v;
}
__device__ __forceinline__ float rowsum16(float v) {
  v += __shfl_xor(v, 1, 32);
  v += __shfl_xor(v, 2, 32);
  v += __shfl_xor(v, 4, 32);
  v += __shfl_xor(v, 8, 32);
  return v;
}

// ---------------- elementwise helpers ----------------

__global__ void f32_to_f16_kernel(const float* __restrict__ src,
                                  _Float16* __restrict__ dst, size_t n) {
  size_t t = (size_t)blockIdx.x * blockDim.x + threadIdx.x;
  if (t < n) dst[t] = (_Float16)src[t];
}

// W_eff[o,i] = NF4[idx]*absmax + 2 * sum_r B[o,r]*A[r,i]   (fp16 out, row-major N x K)
__global__ void dequant_lora_kernel(const int* __restrict__ idx,
                                    const float* __restrict__ am,
                                    const float* __restrict__ A,
                                    const float* __restrict__ Bm,
                                    _Float16* __restrict__ W, int N, int K) {
  size_t t = (size_t)blockIdx.x * blockDim.x + threadIdx.x;
  size_t total = (size_t)N * K;
  if (t >= total) return;
  int o = (int)(t / K);
  int i = (int)(t - (size_t)o * K);
  float w = NF4LUT[idx[t] & 15] * am[t >> 6];
  float l = 0.0f;
#pragma unroll
  for (int r = 0; r < RANK; ++r)
    l = fmaf(Bm[o * RANK + r], A[(size_t)r * K + i], l);
  W[t] = (_Float16)(w + LORA_SCALE * l);
}

// RoPE in place on (S, nheads, 128) fp16; one thread per complex pair
__global__ void rope_kernel(_Float16* __restrict__ q, int nheads,
                            const float* __restrict__ cosv,
                            const float* __restrict__ sinv) {
  size_t t = (size_t)blockIdx.x * blockDim.x + threadIdx.x;
  size_t total = (size_t)S_LEN * nheads * (HD / 2);
  if (t >= total) return;
  int d = (int)(t & 63);
  size_t tmp = t >> 6;
  int h = (int)(tmp % nheads);
  int s = (int)(tmp / nheads);
  _Float16* p = q + ((size_t)s * nheads + h) * HD + 2 * d;
  float xr = (float)p[0], xi = (float)p[1];
  float c = cosv[s * 64 + d], sn = sinv[s * 64 + d];
  p[0] = (_Float16)(xr * c - xi * sn);
  p[1] = (_Float16)(xr * sn + xi * c);
}

// v (S, 1024) -> vt (8, 128, S): vt[(h*128+d)*S + s] = v[s*1024 + h*128 + d]
__global__ void transpose_v_kernel(const _Float16* __restrict__ v,
                                   _Float16* __restrict__ vt) {
  size_t t = (size_t)blockIdx.x * blockDim.x + threadIdx.x;
  size_t total = (size_t)S_LEN * KV_DIM;
  if (t >= total) return;
  int s = (int)(t / KV_DIM);
  int hd = (int)(t - (size_t)s * KV_DIM);
  vt[(size_t)hd * S_LEN + s] = v[t];
}

// ---------------- WMMA GEMM: Y(M,N) = X(M,K) * W(N,K)^T ----------------
// One wave computes a 16x64 tile: 4 accumulators, K-loop step 32.
template <typename OutT>
__global__ __launch_bounds__(256, 1)
void gemm_wmma_kernel(const _Float16* __restrict__ X,
                      const _Float16* __restrict__ W,
                      OutT* __restrict__ Y, int M, int N, int K) {
  int wave = (int)((blockIdx.x * blockDim.x + threadIdx.x) >> 5);
  int lane = threadIdx.x & 31;
  int lm = lane & 15, hi = lane >> 4;
  int nT = N >> 6;
  int mTile = wave / nT;
  int n64 = wave - mTile * nT;
  if (mTile >= (M >> 4)) return;
  int m0 = mTile << 4, n0 = n64 << 6;

  const _Float16* ap  = X + (size_t)(m0 + lm) * K + hi * 8;
  const _Float16* bp0 = W + (size_t)(n0 + lm) * K + hi * 8;

  v8f acc[4] = {v8f{}, v8f{}, v8f{}, v8f{}};
  for (int k = 0; k < K; k += 32) {
    v16h a = cat16(*(const v8h*)(ap + k), *(const v8h*)(ap + k + 16));
#pragma unroll
    for (int j = 0; j < 4; ++j) {
      const _Float16* bp = bp0 + (size_t)j * 16 * K + k;
      v16h b = cat16(*(const v8h*)bp, *(const v8h*)(bp + 16));
      acc[j] = wmma_f16(a, b, acc[j]);
    }
  }
#pragma unroll
  for (int j = 0; j < 4; ++j)
#pragma unroll
    for (int r = 0; r < 8; ++r)
      Y[(size_t)(m0 + r + 8 * hi) * N + n0 + j * 16 + lm] = (OutT)acc[j][r];
}

// ---------------- flash attention ----------------
// One wave: (head h, 16-row query tile). KV step = 32 keys.
// q (S,4096) rope'd, k (S,1024) rope'd, vt (8,128,S), out (S,4096) all fp16.
__global__ __launch_bounds__(256, 1)
void flash_attn_kernel(const _Float16* __restrict__ q,
                       const _Float16* __restrict__ k,
                       const _Float16* __restrict__ vt,
                       _Float16* __restrict__ out) {
  __shared__ _Float16 plds[8 * 16 * 32];  // 1KB per wave, 8 waves
  int wid  = threadIdx.x >> 5;
  int lane = threadIdx.x & 31;
  int lm = lane & 15, hi = lane >> 4;
  int task = (int)(blockIdx.x * 8 + wid);
  if (task >= NH * (S_LEN / 16)) return;
  int qi = task & 127;         // query tile
  int h  = task >> 7;          // head
  int kvh = h >> 2;            // n_rep = 4
  int q0 = qi << 4;
  _Float16* pbase = plds + wid * (16 * 32);

  // Q fragments: 4 chunks of head-dim 32 (kept resident in VGPRs)
  const _Float16* qrow = q + (size_t)(q0 + lm) * D_MODEL + h * HD;
  v16h qf[4];
#pragma unroll
  for (int c = 0; c < 4; ++c) {
    const _Float16* p = qrow + c * 32 + hi * 8;
    qf[c] = cat16(*(const v8h*)p, *(const v8h*)(p + 16));
  }

  const _Float16* kbase  = k  + (size_t)kvh * HD;
  const _Float16* vtbase = vt + (size_t)kvh * HD * S_LEN;

  v8f oacc[8];
#pragma unroll
  for (int c = 0; c < 8; ++c) oacc[c] = v8f{};
  float mrow[8], lrow[8];
#pragma unroll
  for (int r = 0; r < 8; ++r) { mrow[r] = -3.0e38f; lrow[r] = 0.0f; }

  int kv_end = ((q0 + 47) >> 5) << 5;  // cover keys 0..q0+15
  for (int kv0 = 0; kv0 < kv_end; kv0 += 32) {
    // ---- scores: two 16x16 tiles, K over head-dim 128 ----
    v8f s0 = v8f{}, s1 = v8f{};
#pragma unroll
    for (int c = 0; c < 4; ++c) {
      const _Float16* kp0 = kbase + (size_t)(kv0 + lm) * KV_DIM + c * 32 + hi * 8;
      const _Float16* kp1 = kbase + (size_t)(kv0 + 16 + lm) * KV_DIM + c * 32 + hi * 8;
      v16h b0 = cat16(*(const v8h*)kp0, *(const v8h*)(kp0 + 16));
      v16h b1 = cat16(*(const v8h*)kp1, *(const v8h*)(kp1 + 16));
      s0 = wmma_f16(qf[c], b0, s0);
      s1 = wmma_f16(qf[c], b1, s1);
    }
    // ---- mask + online softmax; stage P in LDS as fp16 ----
#pragma unroll
    for (int r = 0; r < 8; ++r) {
      int qr = q0 + r + 8 * hi;
      float a0 = s0[r] * INV_SQRT_HD;
      float a1 = s1[r] * INV_SQRT_HD;
      if (kv0 + lm > qr)      a0 = -3.0e38f;
      if (kv0 + 16 + lm > qr) a1 = -3.0e38f;
      float mt = rowmax16(fmaxf(a0, a1));
      float mn = fmaxf(mrow[r], mt);
      float al = __expf(mrow[r] - mn);
      float p0 = __expf(a0 - mn);
      float p1 = __expf(a1 - mn);
      lrow[r] = lrow[r] * al + rowsum16(p0 + p1);
      mrow[r] = mn;
      pbase[(r + 8 * hi) * 32 + lm]      = (_Float16)p0;
      pbase[(r + 8 * hi) * 32 + 16 + lm] = (_Float16)p1;
#pragma unroll
      for (int c = 0; c < 8; ++c) oacc[c][r] *= al;
    }
    // ---- reload P as A-fragment (16x32) ----
    v16h pf = cat16(*(const v8h*)&pbase[lm * 32 + hi * 8],
                    *(const v8h*)&pbase[lm * 32 + 16 + hi * 8]);
    // ---- PV: 8 head-dim chunks of 16 ----
#pragma unroll
    for (int c = 0; c < 8; ++c) {
      const _Float16* vp = vtbase + (size_t)(c * 16 + lm) * S_LEN + kv0 + hi * 8;
      v16h vb = cat16(*(const v8h*)vp, *(const v8h*)(vp + 16));
      oacc[c] = wmma_f16(pf, vb, oacc[c]);
    }
  }
  // ---- normalize & store ----
#pragma unroll
  for (int c = 0; c < 8; ++c)
#pragma unroll
    for (int r = 0; r < 8; ++r) {
      float val = oacc[c][r] / lrow[r];
      out[(size_t)(q0 + r + 8 * hi) * D_MODEL + h * HD + c * 16 + lm] = (_Float16)val;
    }
}

// ---------------- host launch ----------------
extern "C" void kernel_launch(void* const* d_in, const int* in_sizes, int n_in,
                              void* d_out, int out_size, void* d_ws, size_t ws_size,
                              hipStream_t stream) {
  (void)in_sizes; (void)n_in; (void)out_size; (void)ws_size;
  const float* x    = (const float*)d_in[0];
  const float* cosv = (const float*)d_in[1];
  const float* sinv = (const float*)d_in[2];
  // d_in[3] = mask (causal, reimplemented analytically)

  char* ws = (char*)d_ws;
  _Float16* xh   = (_Float16*)ws; ws += (size_t)S_LEN * D_MODEL * 2;   // 16 MB
  _Float16* wbuf = (_Float16*)ws; ws += (size_t)D_MODEL * D_MODEL * 2; // 32 MB (reused)
  _Float16* qh   = (_Float16*)ws; ws += (size_t)S_LEN * D_MODEL * 2;   // 16 MB
  _Float16* kh   = (_Float16*)ws; ws += (size_t)S_LEN * KV_DIM * 2;    // 4 MB
  _Float16* vh   = (_Float16*)ws; ws += (size_t)S_LEN * KV_DIM * 2;    // 4 MB
  _Float16* vth  = (_Float16*)ws; ws += (size_t)S_LEN * KV_DIM * 2;    // 4 MB
  _Float16* atn  = (_Float16*)ws; ws += (size_t)S_LEN * D_MODEL * 2;   // 16 MB

  // x -> fp16
  {
    size_t n = (size_t)S_LEN * D_MODEL;
    f32_to_f16_kernel<<<(unsigned)((n + 255) / 256), 256, 0, stream>>>(x, xh, n);
  }
  auto dequant = [&](int base, _Float16* W, int N) {
    size_t total = (size_t)N * D_MODEL;
    dequant_lora_kernel<<<(unsigned)((total + 255) / 256), 256, 0, stream>>>(
        (const int*)d_in[base], (const float*)d_in[base + 1],
        (const float*)d_in[base + 2], (const float*)d_in[base + 3], W, N, D_MODEL);
  };
  auto gemmBlocks = [](int M, int N) {
    return (unsigned)(((size_t)(M / 16) * (N / 64) + 7) / 8);
  };

  // Q projection
  dequant(4, wbuf, D_MODEL);
  gemm_wmma_kernel<_Float16><<<gemmBlocks(S_LEN, D_MODEL), 256, 0, stream>>>(
      xh, wbuf, qh, S_LEN, D_MODEL, D_MODEL);
  // K projection
  dequant(8, wbuf, KV_DIM);
  gemm_wmma_kernel<_Float16><<<gemmBlocks(S_LEN, KV_DIM), 256, 0, stream>>>(
      xh, wbuf, kh, S_LEN, KV_DIM, D_MODEL);
  // V projection
  dequant(12, wbuf, KV_DIM);
  gemm_wmma_kernel<_Float16><<<gemmBlocks(S_LEN, KV_DIM), 256, 0, stream>>>(
      xh, wbuf, vh, S_LEN, KV_DIM, D_MODEL);

  // RoPE
  {
    size_t nq = (size_t)S_LEN * NH * (HD / 2);
    rope_kernel<<<(unsigned)((nq + 255) / 256), 256, 0, stream>>>(qh, NH, cosv, sinv);
    size_t nk = (size_t)S_LEN * NKVH * (HD / 2);
    rope_kernel<<<(unsigned)((nk + 255) / 256), 256, 0, stream>>>(kh, NKVH, cosv, sinv);
  }
  // transpose V
  {
    size_t n = (size_t)S_LEN * KV_DIM;
    transpose_v_kernel<<<(unsigned)((n + 255) / 256), 256, 0, stream>>>(vh, vth);
  }
  // attention: 32 heads * 128 q-tiles = 4096 waves -> 512 blocks
  flash_attn_kernel<<<512, 256, 0, stream>>>(qh, kh, vth, atn);

  // O projection (f32 out)
  dequant(16, wbuf, D_MODEL);
  gemm_wmma_kernel<float><<<gemmBlocks(S_LEN, D_MODEL), 256, 0, stream>>>(
      atn, wbuf, (float*)d_out, S_LEN, D_MODEL, D_MODEL);
}
